// NoiseRecipe_25013889532586
// MI455X (gfx1250) — matrix operand order repaired and checked by reading, not compiled, not run on hardware
//
#include <hip/hip_runtime.h>
#include <hip/hip_bf16.h>

#define B_ROWS  2048
#define N_ITEMS 100000
#define DIM     64
#define K_HARD  5
#define K_LIST  6   // keep top-6; positive item (self-sim ~1.0) removed at merge

typedef __attribute__((ext_vector_type(16))) __bf16 v16bf;
typedef __attribute__((ext_vector_type(8)))  __bf16 v8bf;
typedef __attribute__((ext_vector_type(8)))  float  v8f;

// Unguarded branchless compare-exchange insertion into descending top-K_LIST.
// One v_cmp + 4 v_cndmask per stage (selects reuse the compare; avoids the
// canonicalizing v_max_num(v,v) ops that fmaxf/fminf emit).
__device__ __forceinline__ void sort_insert(float v, int id,
                                            float tv[K_LIST], int ti[K_LIST]) {
#pragma unroll
    for (int j = 0; j < K_LIST; ++j) {
        bool  c  = v > tv[j];
        float vh = c ? v : tv[j];
        float vl = c ? tv[j] : v;
        int   dh = c ? id : ti[j];
        int   dl = c ? ti[j] : id;
        tv[j] = vh; ti[j] = dh; v = vl; id = dl;
    }
}

// Pass 1: per-item inverse L2 norm (matches F.normalize: 1/max(||x||, eps)).
__global__ void item_invnorm_kernel(const float* __restrict__ items,
                                    float* __restrict__ inv_norm) {
    int i = blockIdx.x * blockDim.x + threadIdx.x;
    if (i >= N_ITEMS) return;
    const float4* p = (const float4*)(items + (size_t)i * DIM);
    float s = 0.f;
#pragma unroll
    for (int j = 0; j < DIM / 4; ++j) {
        float4 v = p[j];
        s += v.x * v.x + v.y * v.y + v.z * v.z + v.w * v.w;
    }
    inv_norm[i] = 1.0f / fmaxf(sqrtf(s), 1e-12f);
}

// Pass 1b (optional): normalized bf16 item table, row-major [N_ITEMS x DIM].
__global__ void item_bf16_kernel(const float* __restrict__ items,
                                 const float* __restrict__ inv_norm,
                                 __bf16* __restrict__ itemsBf) {
    int t = blockIdx.x * blockDim.x + threadIdx.x;
    if (t >= N_ITEMS * (DIM / 8)) return;
    int i = t >> 3, q = t & 7;
    float s = inv_norm[i];
    const float4* p = (const float4*)(items + (size_t)i * DIM + q * 8);
    float4 a = p[0], b = p[1];
    v8bf o;
    o[0] = (__bf16)(a.x * s); o[1] = (__bf16)(a.y * s);
    o[2] = (__bf16)(a.z * s); o[3] = (__bf16)(a.w * s);
    o[4] = (__bf16)(b.x * s); o[5] = (__bf16)(b.y * s);
    o[6] = (__bf16)(b.z * s); o[7] = (__bf16)(b.w * s);
    *(v8bf*)(itemsBf + (size_t)i * DIM + q * 8) = o;
}

// Pass 2: gather + normalize positive rows into workspace [B_ROWS x DIM] f32.
__global__ void gather_pos_kernel(const float* __restrict__ items,
                                  const float* __restrict__ inv_norm,
                                  const int* __restrict__ pos_items,
                                  float* __restrict__ posAn) {
    int t = blockIdx.x * blockDim.x + threadIdx.x;
    if (t >= B_ROWS * (DIM / 4)) return;
    int b = t >> 4, q = t & 15;
    int it = pos_items[b];
    float s = inv_norm[it];
    float4 v = ((const float4*)(items + (size_t)it * DIM))[q];
    float4 o; o.x = v.x * s; o.y = v.y * s; o.z = v.z * s; o.w = v.w * s;
    ((float4*)(posAn + (size_t)b * DIM))[q] = o;
}

#define PK4(dst, off, v)                        \
    dst[(off) + 0] = (__bf16)((v).x * sc);      \
    dst[(off) + 1] = (__bf16)((v).y * sc);      \
    dst[(off) + 2] = (__bf16)((v).z * sc);      \
    dst[(off) + 3] = (__bf16)((v).w * sc);

// Pass 3: fused bf16-WMMA GEMM + per-row top-6 + edge emission.
// One workgroup = 16 batch rows; 8 waves stride the 6250 item tiles.
// A = items (M), B = pos rows (N): each lane's 8 accumulator values all
// belong to ONE batch row -> pure per-lane top-k in the hot loop.
template <bool PRESCALED>
__global__ void __launch_bounds__(256) simtopk_kernel(
    const float* __restrict__ items, const float* __restrict__ inv_norm,
    const __bf16* __restrict__ itemsBf, const float* __restrict__ posAn,
    const int* __restrict__ users, const int* __restrict__ pos_items,
    float* __restrict__ out) {
    __shared__ float s_val[256 * K_LIST];
    __shared__ int   s_idx[256 * K_LIST];

    const int tid  = threadIdx.x;
    const int lane = tid & 31;
    const int wave = tid >> 5;
    const int hi   = lane >> 4;        // half-lane selector
    const int nrow = lane & 15;        // batch row within tile (B/C N index)
    const int r0   = blockIdx.x * 16;
    const int row  = r0 + nrow;

    // --- B fragments (normalized pos row), bf16, built once ---
    const float* prow = posAn + (size_t)row * DIM + hi * 8;
    v16bf b_lo, b_hi;
    {
        const float sc = 1.0f;  // already normalized
#pragma unroll
        for (int g = 0; g < 4; ++g) { float4 v = ((const float4*)prow)[g + (g >> 1) * 2];        PK4(b_lo, 4 * g, v) }
#pragma unroll
        for (int g = 0; g < 4; ++g) { float4 v = ((const float4*)(prow + 32))[g + (g >> 1) * 2]; PK4(b_hi, 4 * g, v) }
    }

    float tv[K_LIST];
    int   ti[K_LIST];
#pragma unroll
    for (int j = 0; j < K_LIST; ++j) { tv[j] = -3.0f; ti[j] = -1; }

#pragma unroll 2
    for (int m0 = wave * 16; m0 < N_ITEMS; m0 += 16 * 8) {
        const int mitem = m0 + nrow;                     // A-matrix row M
        // m0 is wave-uniform; readfirstlane makes the prefetch guard scalar
        // (s_cmp + s_cbranch instead of a v_cmpx exec diamond).
        const bool pf = __builtin_amdgcn_readfirstlane(m0) + 16 * 8 + 15 < N_ITEMS;
        v16bf a_lo, a_hi;
        if (PRESCALED) {
            const __bf16* arow = itemsBf + (size_t)mitem * DIM + hi * 8;
            if (pf) __builtin_prefetch(arow + 128 * DIM, 0, 1);
            v8bf g0 = *(const v8bf*)(arow +  0);
            v8bf g1 = *(const v8bf*)(arow + 16);
            v8bf g2 = *(const v8bf*)(arow + 32);
            v8bf g3 = *(const v8bf*)(arow + 48);
            a_lo = __builtin_shufflevector(g0, g1, 0,1,2,3,4,5,6,7,8,9,10,11,12,13,14,15);
            a_hi = __builtin_shufflevector(g2, g3, 0,1,2,3,4,5,6,7,8,9,10,11,12,13,14,15);
        } else {
            const float sc    = inv_norm[mitem];
            const float* arow = items + (size_t)mitem * DIM + hi * 8;
            if (pf) __builtin_prefetch(arow + 128 * DIM, 0, 1);
            float4 f0 = ((const float4*)(arow +  0))[0];
            float4 f1 = ((const float4*)(arow +  4))[0];
            float4 f2 = ((const float4*)(arow + 16))[0];
            float4 f3 = ((const float4*)(arow + 20))[0];
            float4 f4 = ((const float4*)(arow + 32))[0];
            float4 f5 = ((const float4*)(arow + 36))[0];
            float4 f6 = ((const float4*)(arow + 48))[0];
            float4 f7 = ((const float4*)(arow + 52))[0];
            PK4(a_lo,  0, f0) PK4(a_lo,  4, f1) PK4(a_lo,  8, f2) PK4(a_lo, 12, f3)
            PK4(a_hi,  0, f4) PK4(a_hi,  4, f5) PK4(a_hi,  8, f6) PK4(a_hi, 12, f7)
        }

        v8f c = {};
        c = __builtin_amdgcn_wmma_f32_16x16x32_bf16(false, a_lo, false, b_lo,
                                                    (short)0, c, false, false);
        c = __builtin_amdgcn_wmma_f32_16x16x32_bf16(false, a_hi, false, b_hi,
                                                    (short)0, c, false, false);

        // Single group guard per tile: max-reduce the 8 candidates (lowers to
        // v_max3_num_f32 tree) vs the running 6th-best; cascades are rare.
        float gmax = fmaxf(fmaxf(fmaxf(c[0], c[1]), fmaxf(c[2], c[3])),
                           fmaxf(fmaxf(c[4], c[5]), fmaxf(c[6], c[7])));
        if (gmax > tv[K_LIST - 1]) {
            const int mbase = m0 + hi * 8;               // C layout: M = mbase + j
#pragma unroll
            for (int j = 0; j < 8; ++j)
                sort_insert(c[j], mbase + j, tv, ti);
        }
    }

    // --- merge: 16 per-lane lists -> per-row top-6, then drop positive ---
#pragma unroll
    for (int j = 0; j < K_LIST; ++j) {
        s_val[tid * K_LIST + j] = tv[j];
        s_idx[tid * K_LIST + j] = ti[j];
    }
    __syncthreads();

    if (tid < 16) {
        float fv[K_LIST]; int fx[K_LIST];
#pragma unroll
        for (int j = 0; j < K_LIST; ++j) { fv[j] = -3.0f; fx[j] = -1; }
        for (int w = 0; w < 8; ++w) {
#pragma unroll
            for (int h = 0; h < 2; ++h) {
                int l = w * 32 + h * 16 + tid;
#pragma unroll
                for (int j = 0; j < K_LIST; ++j)
                    sort_insert(s_val[l * K_LIST + j], s_idx[l * K_LIST + j], fv, fx);
            }
        }
        int   orow = r0 + tid;
        int   mp   = pos_items[orow];
        float uu   = (float)users[orow];
        out[orow]          = uu;                          // edge_pos_u
        out[B_ROWS + orow] = (float)mp;                   // edge_pos_i
        float* neg_u = out + 2 * B_ROWS;
        float* neg_i = out + 2 * B_ROWS + B_ROWS * K_HARD;
        int cnt = 0;
        for (int j = 0; j < K_LIST && cnt < K_HARD; ++j) {
            if (fx[j] != mp) {                            // scatter-mask equivalent
                neg_u[orow * K_HARD + cnt] = uu;          // edge_neg_u
                neg_i[orow * K_HARD + cnt] = (float)fx[j];// edge_neg_i
                ++cnt;
            }
        }
    }
}

extern "C" void kernel_launch(void* const* d_in, const int* in_sizes, int n_in,
                              void* d_out, int out_size, void* d_ws, size_t ws_size,
                              hipStream_t stream) {
    (void)in_sizes; (void)n_in; (void)out_size;
    const int*   users     = (const int*)d_in[0];
    const int*   pos_items = (const int*)d_in[1];
    // d_in[2] = neg_items: unused by the reference outputs
    const float* items     = (const float*)d_in[3];
    // d_in[4] = k_hard (== 5, hardcoded)

    float*  ws       = (float*)d_ws;
    float*  inv_norm = ws;                 // 100352 floats
    float*  posAn    = ws + 100352;        // 131072 floats
    __bf16* itemsBf  = (__bf16*)(ws + 100352 + 131072);  // 6.4M bf16 (12.8 MB)
    float*  out      = (float*)d_out;

    const size_t need_bf =
        (size_t)(100352 + 131072) * sizeof(float) + (size_t)N_ITEMS * DIM * sizeof(__bf16);

    item_invnorm_kernel<<<(N_ITEMS + 255) / 256, 256, 0, stream>>>(items, inv_norm);
    gather_pos_kernel<<<(B_ROWS * (DIM / 4) + 255) / 256, 256, 0, stream>>>(
        items, inv_norm, pos_items, posAn);

    if (ws_size >= need_bf) {
        item_bf16_kernel<<<(N_ITEMS * (DIM / 8) + 255) / 256, 256, 0, stream>>>(
            items, inv_norm, itemsBf);
        simtopk_kernel<true><<<B_ROWS / 16, 256, 0, stream>>>(
            items, inv_norm, itemsBf, posAn, users, pos_items, out);
    } else {
        simtopk_kernel<false><<<B_ROWS / 16, 256, 0, stream>>>(
            items, inv_norm, itemsBf, posAn, users, pos_items, out);
    }
}
